// GroupedQueryAttention_19705309954535
// MI455X (gfx1250) — matrix operand order repaired and checked by reading, not compiled, not run on hardware
//
#include <hip/hip_runtime.h>
#include <stdint.h>

// Problem constants (from reference)
#define B_  2
#define S_  2048
#define E_  2048
#define H_  16
#define KV_ 4
#define D_  128
#define G_  (H_ / KV_)
#define M_  (B_ * S_)          // 4096 total rows (b*s)
#define THETA_ 10000.0f

typedef __bf16 v16bf __attribute__((ext_vector_type(16)));
typedef float  v8f   __attribute__((ext_vector_type(8)));
typedef int    vi4   __attribute__((vector_size(16)));   // b128 payload type

// CDNA5 async global->LDS path (ASYNCcnt), guarded so both toolchains compile.
#if defined(__HIP_DEVICE_COMPILE__) && \
    __has_builtin(__builtin_amdgcn_global_load_async_to_lds_b128) && \
    __has_builtin(__builtin_amdgcn_s_wait_asynccnt)
#define USE_ASYNC 1
#define GP128(p) ((__attribute__((address_space(1))) vi4*)(p))
#define LP128(p) ((__attribute__((address_space(3))) vi4*)(p))
#else
#define USE_ASYNC 0
#endif

__device__ __forceinline__ void async_wait0() {
#if USE_ASYNC
  __builtin_amdgcn_s_wait_asynccnt(0);
#endif
}

// copy 32 bytes global->LDS per lane
__device__ __forceinline__ void cp32(const __bf16* g, __bf16* l) {
#if USE_ASYNC
  __builtin_amdgcn_global_load_async_to_lds_b128(GP128(g), LP128(l), 0, 0);
  __builtin_amdgcn_global_load_async_to_lds_b128(GP128(g + 8), LP128(l + 8), 0, 0);
#else
  ((uint4*)l)[0] = ((const uint4*)g)[0];
  ((uint4*)l)[1] = ((const uint4*)g)[1];
#endif
}
// copy 128 bytes global->LDS per lane
__device__ __forceinline__ void cp128(const __bf16* g, __bf16* l) {
#if USE_ASYNC
#pragma unroll
  for (int i = 0; i < 8; ++i)
    __builtin_amdgcn_global_load_async_to_lds_b128(GP128(g + i * 8),
                                                   LP128(l + i * 8), 0, 0);
#else
#pragma unroll
  for (int i = 0; i < 8; ++i) ((uint4*)l)[i] = ((const uint4*)g)[i];
#endif
}

// ---------- bf16 <-> f32 helpers (RNE) ----------
__device__ __forceinline__ __bf16 f2bf(float f) {
  uint32_t u = __builtin_bit_cast(uint32_t, f);
  uint32_t r = u + 0x7FFFu + ((u >> 16) & 1u);
  uint16_t h = (uint16_t)(r >> 16);
  return __builtin_bit_cast(__bf16, h);
}
__device__ __forceinline__ float bf2f(__bf16 b) {
  uint16_t h = __builtin_bit_cast(uint16_t, b);
  uint32_t u = ((uint32_t)h) << 16;
  return __builtin_bit_cast(float, u);
}
__device__ __forceinline__ v8f vzero8() {
  v8f z;
#pragma unroll
  for (int i = 0; i < 8; ++i) z[i] = 0.0f;
  return z;
}

// ---------- WMMA wrapper: D = A(16x32 bf16) * B(32x16 bf16) + C(f32) ----------
__device__ __forceinline__ v8f wmma_bf16(v16bf a, v16bf b, v8f c) {
  return __builtin_amdgcn_wmma_f32_16x16x32_bf16(
      false, a, false, b, (short)0, c, false, false);
}

// ---------- Fragment loaders, contiguous k order (k = 16*(lane/16) + i) in both A and B ----------
// A (16Mx32K), row-major source (ld elements): one contiguous 32B read/lane.
__device__ __forceinline__ v16bf load_A_frag(const __bf16* a, int ld) {
  int lane = threadIdx.x & 31;
  int row  = lane & 15;
  int kb   = (lane >> 4) * 16;  // 0 or 16
  const __bf16* p = a + (size_t)row * ld;
  v16bf f;
#pragma unroll
  for (int i = 0; i < 16; ++i) f[i] = p[kb + i];
  return f;
}
// B (32Kx16N) fragment from an N-major tile: element(K=k,N=n) = t[n*ld + k].
// One contiguous 32B read per lane -> 2x ds_load_b128.
__device__ __forceinline__ v16bf load_Bt_frag(const __bf16* t, int ld) {
  int lane = threadIdx.x & 31;
  int col  = lane & 15;
  int kb   = (lane >> 4) * 16;  // 0 or 16
  const __bf16* p = t + (size_t)col * ld + kb;
  v16bf f;
#pragma unroll
  for (int i = 0; i < 16; ++i) f[i] = p[i];
  return f;
}
// C/D layout: element r of v8f -> row = r + 8*(lane>=16), col = lane&15
__device__ __forceinline__ void cstore(__bf16* p, float v) { *p = f2bf(v); }
__device__ __forceinline__ void cstore(float* p, float v)  { *p = v; }

// ---------- fp32 -> bf16 convert (straight) ----------
__global__ __launch_bounds__(256) void cvt_f32_bf16(const float* __restrict__ in,
                                                    __bf16* __restrict__ out, int n) {
  int i = blockIdx.x * 256 + threadIdx.x;
  if (i < n) out[i] = f2bf(in[i]);
}
// ---------- fp32 (KxN) -> bf16 transposed (NxK), LDS-tiled for coalescing ----------
__global__ __launch_bounds__(256) void cvt_f32_bf16_T(const float* __restrict__ in,
                                                      __bf16* __restrict__ out,
                                                      int K, int N) {
  __shared__ float tile[32][33];
  int n0 = blockIdx.x * 32, k0 = blockIdx.y * 32;
  int tx = threadIdx.x & 31, ty = threadIdx.x >> 5;   // 32 x 8
#pragma unroll
  for (int r = 0; r < 32; r += 8)
    tile[r + ty][tx] = in[(size_t)(k0 + r + ty) * N + n0 + tx];
  __syncthreads();
#pragma unroll
  for (int r = 0; r < 32; r += 8)
    out[(size_t)(n0 + r + ty) * K + k0 + tx] = f2bf(tile[tx][r + ty]);
}

// ---------- bf16 WMMA GEMM: C(MxN) = A(MxK) * Bt(NxK)^T, double-buffered LDS ----
// Block: 256 threads = 8 waves; tile 128x128x32. Wave (wm,wn) does 32x64.
#define BM 128
#define BN 128
#define BK 32
template <typename OutT>
__global__ __launch_bounds__(256) void gemm_bf16(const __bf16* __restrict__ A,
                                                 const __bf16* __restrict__ Bt,
                                                 OutT* __restrict__ C,
                                                 int M, int N, int K) {
  __shared__ __bf16 As[2][BM * BK];    // 2 x 8 KB
  __shared__ __bf16 BsT[2][BN * BK];   // 2 x 8 KB (N-major)
  const int bn = blockIdx.x, bm = blockIdx.y;
  const int tid = threadIdx.x;
  const int wave = tid >> 5;
  const int wm = wave & 3;    // 4 row-groups of 32
  const int wn = wave >> 2;   // 2 col-groups of 64
  const int srow = tid >> 1, shalf = tid & 1;   // staging: 128 rows x 2 halves

  v8f acc[2][4];
#pragma unroll
  for (int i = 0; i < 2; ++i)
#pragma unroll
    for (int j = 0; j < 4; ++j) acc[i][j] = vzero8();

  auto stage = [&](int bb, int k0) {
    cp32(A  + (size_t)(bm * BM + srow) * K + k0 + shalf * 16,
         &As[bb][srow * BK + shalf * 16]);
    cp32(Bt + (size_t)(bn * BN + srow) * K + k0 + shalf * 16,
         &BsT[bb][srow * BK + shalf * 16]);
  };

  stage(0, 0);
  async_wait0();
  __syncthreads();

  int buf = 0;
  for (int k0 = 0; k0 < K; k0 += BK) {
    if (k0 + BK < K) stage(buf ^ 1, k0 + BK);   // overlap with WMMAs below

    v16bf af[2], bfr[4];
#pragma unroll
    for (int ms = 0; ms < 2; ++ms)
      af[ms] = load_A_frag(&As[buf][(wm * 32 + ms * 16) * BK], BK);
#pragma unroll
    for (int ns = 0; ns < 4; ++ns)
      bfr[ns] = load_Bt_frag(&BsT[buf][(wn * 64 + ns * 16) * BK], BK);
#pragma unroll
    for (int ms = 0; ms < 2; ++ms)
#pragma unroll
      for (int ns = 0; ns < 4; ++ns)
        acc[ms][ns] = wmma_bf16(af[ms], bfr[ns], acc[ms][ns]);

    async_wait0();
    __syncthreads();
    buf ^= 1;
  }

  // Epilogue
  const int lane = tid & 31;
  const int col = lane & 15;
  const int rb = (lane >> 4) * 8;
#pragma unroll
  for (int ms = 0; ms < 2; ++ms)
#pragma unroll
    for (int ns = 0; ns < 4; ++ns) {
      OutT* c0 = C + (size_t)(bm * BM + wm * 32 + ms * 16 + rb) * N
                   + bn * BN + wn * 64 + ns * 16 + col;
#pragma unroll
      for (int r = 0; r < 8; ++r) cstore(c0 + (size_t)r * N, acc[ms][ns][r]);
    }
}

// ---------- RoPE (in place, bf16 storage, fp32 math) ----------
__global__ __launch_bounds__(256) void rope_kernel(__bf16* __restrict__ X,
                                                   int rows, int heads) {
  int idx = blockIdx.x * 256 + threadIdx.x;
  int total = rows * heads * (D_ / 2);
  if (idx >= total) return;
  int i = idx & 63;                    // pair index within half-dim (D/2=64)
  int h = (idx >> 6) % heads;
  int row = idx / (64 * heads);
  int pos = row % S_;                  // sequence position
  float inv = __powf(THETA_, -(float)(2 * i) / (float)D_);
  float ang = (float)pos * inv;
  float c = __cosf(ang), sn = __sinf(ang);
  __bf16* p = X + (size_t)row * (heads * D_) + h * D_;
  float x1 = bf2f(p[i]), x2 = bf2f(p[i + 64]);
  p[i]      = f2bf(x1 * c - x2 * sn);
  p[i + 64] = f2bf(x2 * c + x1 * sn);
}

// ---------- Flash attention (causal, GQA), 64 queries/block, 4 waves ----------
__global__ __launch_bounds__(128) void attn_kernel(const __bf16* __restrict__ Q,
                                                   const __bf16* __restrict__ K,
                                                   const __bf16* __restrict__ V,
                                                   __bf16* __restrict__ O) {
  __shared__ __bf16 Ks[64 * 128];      // 16 KB, [key][d]  (also Q staging)
  __shared__ __bf16 Vst[128 * 64];     // 16 KB, [d][key]  (transposed V)
  __shared__ __bf16 Ps[4][16 * 64];    // 8 KB, per-wave P tile

  const int qt = blockIdx.x;           // query tile (64 rows)
  const int h  = blockIdx.y;           // query head
  const int b  = blockIdx.z;
  const int kv = h / G_;               // shared KV head
  const int tid = threadIdx.x;
  const int w = tid >> 5;              // wave -> owns q rows [w*16, w*16+16)
  const int lane = tid & 31;
  const int coln = lane & 15;
  const int rb = (lane >> 4) * 8;      // C-frag row base for this lane
  const float scale = 0.08838834764831845f;   // 1/sqrt(128)

  const int row2 = tid >> 1, half = tid & 1;  // staging: 64 rows x 2 halves

  // Stage Q tile (64x128) via Ks, then pull per-wave A-fragments into regs
  cp128(Q + (size_t)(b * S_ + qt * 64 + row2) * (H_ * D_) + h * D_ + half * 64,
        Ks + row2 * 128 + half * 64);
  async_wait0();
  __syncthreads();
  v16bf qf[4];
#pragma unroll
  for (int kc = 0; kc < 4; ++kc)
    qf[kc] = load_A_frag(Ks + (size_t)(w * 16) * 128 + kc * 32, 128);
  __syncthreads();

  // Online softmax state (8 rows per lane) + output accumulators (16x128)
  float m[8], l[8];
#pragma unroll
  for (int r = 0; r < 8; ++r) { m[r] = -__builtin_inff(); l[r] = 0.0f; }
  v8f o[8];
#pragma unroll
  for (int dn = 0; dn < 8; ++dn) o[dn] = vzero8();

  for (int j = 0; j <= qt; ++j) {
    // Stage K tile (async, [key][d]) and V tile (manual transpose to [d][key])
    cp128(K + (size_t)(b * S_ + j * 64 + row2) * (KV_ * D_) + kv * D_ + half * 64,
          Ks + row2 * 128 + half * 64);
    {
      const __bf16* vsrc = V + (size_t)(b * S_ + j * 64 + row2) * (KV_ * D_)
                             + kv * D_ + half * 64;
      uint4 tmp[8];
#pragma unroll
      for (int i = 0; i < 8; ++i) tmp[i] = ((const uint4*)vsrc)[i];
      const __bf16* tb = (const __bf16*)tmp;
#pragma unroll
      for (int d = 0; d < 64; ++d)
        Vst[(size_t)(half * 64 + d) * 64 + row2] = tb[d];
    }
    async_wait0();
    __syncthreads();

    // S = Q * K^T  (4 key sub-tiles of 16, contraction over D=128 in 4 chunks)
    v8f s[4];
#pragma unroll
    for (int t = 0; t < 4; ++t) {
      s[t] = vzero8();
#pragma unroll
      for (int kc = 0; kc < 4; ++kc) {
        v16bf kb = load_Bt_frag(Ks + (size_t)(t * 16) * 128 + kc * 32, 128);
        s[t] = wmma_bf16(qf[kc], kb, s[t]);
      }
    }

    // scale + causal mask (diagonal block only)
#pragma unroll
    for (int t = 0; t < 4; ++t)
#pragma unroll
      for (int r = 0; r < 8; ++r) {
        float sc = s[t][r] * scale;
        if (j == qt) {
          int kidx = j * 64 + t * 16 + coln;
          int qidx = qt * 64 + w * 16 + rb + r;
          if (kidx > qidx) sc = -__builtin_inff();
        }
        s[t][r] = sc;
      }

    // row max across 4 tiles + 16 lanes sharing each row
    float mj[8];
#pragma unroll
    for (int r = 0; r < 8; ++r) {
      float v0 = s[0][r];
#pragma unroll
      for (int t = 1; t < 4; ++t) v0 = fmaxf(v0, s[t][r]);
#pragma unroll
      for (int mk = 1; mk < 16; mk <<= 1) v0 = fmaxf(v0, __shfl_xor(v0, mk, 32));
      mj[r] = v0;
    }

    float alpha[8], mnew[8], rsum[8];
#pragma unroll
    for (int r = 0; r < 8; ++r) {
      mnew[r] = fmaxf(m[r], mj[r]);
      alpha[r] = __expf(m[r] - mnew[r]);
      m[r] = mnew[r];
      rsum[r] = 0.0f;
    }

    // P = exp(S - m), stash as bf16 in per-wave LDS tile (row-major 16x64)
#pragma unroll
    for (int t = 0; t < 4; ++t)
#pragma unroll
      for (int r = 0; r < 8; ++r) {
        float p = __expf(s[t][r] - mnew[r]);
        rsum[r] += p;
        Ps[w][(size_t)(rb + r) * 64 + t * 16 + coln] = f2bf(p);
      }
#pragma unroll
    for (int r = 0; r < 8; ++r) {
#pragma unroll
      for (int mk = 1; mk < 16; mk <<= 1) rsum[r] += __shfl_xor(rsum[r], mk, 32);
      l[r] = alpha[r] * l[r] + rsum[r];
    }
    // rescale O accumulators
#pragma unroll
    for (int dn = 0; dn < 8; ++dn)
#pragma unroll
      for (int r = 0; r < 8; ++r) o[dn][r] *= alpha[r];

    // O += P(16x64) * V(64x128); V fragments contiguous from transposed tile
    v16bf pa[2];
#pragma unroll
    for (int c2 = 0; c2 < 2; ++c2) pa[c2] = load_A_frag(&Ps[w][c2 * 32], 64);
#pragma unroll
    for (int dn = 0; dn < 8; ++dn)
#pragma unroll
      for (int c2 = 0; c2 < 2; ++c2) {
        v16bf vb = load_Bt_frag(Vst + (size_t)(dn * 16) * 64 + c2 * 32, 64);
        o[dn] = wmma_bf16(pa[c2], vb, o[dn]);
      }
    __syncthreads();
  }

  // Normalize and write (b, s, h, d) as bf16
  float linv[8];
#pragma unroll
  for (int r = 0; r < 8; ++r) linv[r] = 1.0f / l[r];
#pragma unroll
  for (int dn = 0; dn < 8; ++dn)
#pragma unroll
    for (int r = 0; r < 8; ++r) {
      size_t orow = (size_t)(b * S_ + qt * 64 + w * 16 + rb + r);
      O[orow * (H_ * D_) + h * D_ + dn * 16 + coln] = f2bf(o[dn][r] * linv[r]);
    }
}

// ---------- host-side launch ----------
extern "C" void kernel_launch(void* const* d_in, const int* in_sizes, int n_in,
                              void* d_out, int out_size, void* d_ws, size_t ws_size,
                              hipStream_t stream) {
  (void)in_sizes; (void)n_in; (void)out_size; (void)ws_size;
  const float* x  = (const float*)d_in[0];
  const float* Wq = (const float*)d_in[1];
  const float* Wk = (const float*)d_in[2];
  const float* Wv = (const float*)d_in[3];
  const float* Wo = (const float*)d_in[4];
  float* out = (float*)d_out;

  // Workspace layout (bf16 buffers), 256B aligned
  char* ws = (char*)d_ws;
  size_t off = 0;
  auto alloc = [&](size_t elems) -> __bf16* {
    off = (off + 255) & ~(size_t)255;
    __bf16* p = (__bf16*)(ws + off);
    off += elems * sizeof(__bf16);
    return p;
  };
  __bf16* xb  = alloc((size_t)M_ * E_);
  __bf16* Wqt = alloc((size_t)E_ * H_ * D_);    // transposed (N x K)
  __bf16* Wkt = alloc((size_t)E_ * KV_ * D_);
  __bf16* Wvt = alloc((size_t)E_ * KV_ * D_);
  __bf16* Wot = alloc((size_t)E_ * E_);
  __bf16* qb  = alloc((size_t)M_ * H_ * D_);
  __bf16* kb  = alloc((size_t)M_ * KV_ * D_);
  __bf16* vb  = alloc((size_t)M_ * KV_ * D_);
  __bf16* ao  = alloc((size_t)M_ * H_ * D_);

  // Convert x straight; convert weights with transpose (K x N -> N x K)
  cvt_f32_bf16<<<(M_ * E_ + 255) / 256, 256, 0, stream>>>(x, xb, M_ * E_);
  cvt_f32_bf16_T<<<dim3((H_ * D_) / 32, E_ / 32), 256, 0, stream>>>(Wq, Wqt, E_, H_ * D_);
  cvt_f32_bf16_T<<<dim3((KV_ * D_) / 32, E_ / 32), 256, 0, stream>>>(Wk, Wkt, E_, KV_ * D_);
  cvt_f32_bf16_T<<<dim3((KV_ * D_) / 32, E_ / 32), 256, 0, stream>>>(Wv, Wvt, E_, KV_ * D_);
  cvt_f32_bf16_T<<<dim3(E_ / 32, E_ / 32), 256, 0, stream>>>(Wo, Wot, E_, E_);

  // q/k/v projections (bf16 WMMA, fp32 accumulate, bf16 out)
  gemm_bf16<__bf16><<<dim3((H_ * D_) / BN, M_ / BM), 256, 0, stream>>>(
      xb, Wqt, qb, M_, H_ * D_, E_);
  gemm_bf16<__bf16><<<dim3((KV_ * D_) / BN, M_ / BM), 256, 0, stream>>>(
      xb, Wkt, kb, M_, KV_ * D_, E_);
  gemm_bf16<__bf16><<<dim3((KV_ * D_) / BN, M_ / BM), 256, 0, stream>>>(
      xb, Wvt, vb, M_, KV_ * D_, E_);

  // RoPE on q and k
  {
    int nq = M_ * H_ * (D_ / 2);
    rope_kernel<<<(nq + 255) / 256, 256, 0, stream>>>(qb, M_, H_);
    int nk = M_ * KV_ * (D_ / 2);
    rope_kernel<<<(nk + 255) / 256, 256, 0, stream>>>(kb, M_, KV_);
  }

  // causal GQA flash attention
  attn_kernel<<<dim3(S_ / 64, H_, B_), 128, 0, stream>>>(qb, kb, vb, ao);

  // output projection -> fp32 d_out
  gemm_bf16<float><<<dim3(E_ / BN, M_ / BM), 256, 0, stream>>>(
      ao, Wot, out, M_, E_, E_);
}
